// GATRelateCNetV2_22101901705655
// MI455X (gfx1250) — matrix operand order repaired and checked by reading, not compiled, not run on hardware
//
#include <hip/hip_runtime.h>
#include <hip/hip_bf16.h>
#include <math.h>

#define BB    16
#define CIN0  16
#define NIND  300
#define SS    128
#define OUTC  32
#define GCNC  4
#define NLAY  3
#define EDGES 153600
#define NN    (BB * NIND)          // 4800
#define HALFN (NIND / 2)           // 150
#define FEAT  (GCNC * SS)          // 512
#define CMID  (OUTC + GCNC)        // 36

typedef _Float16 v16h __attribute__((ext_vector_type(16)));
typedef _Float16 v8h  __attribute__((ext_vector_type(8)));
typedef float    v8f  __attribute__((ext_vector_type(8)));

// ---------------------------------------------------------------------------
// Conv (1x3), pad (0,1), via WMMA f32<=f16. One block (128 thr) per (b,n) row.
// D[o][s] = sum_k A[o][k] * Bm[k][s],  k = c*3 + t,  Bm[k][s] = in[b][c][n][s+t-1]
// LDS holds A ([o][k] linear) and an im2col-transposed B ([s][k], padded row
// stride) so fragments are wide contiguous ds_load_b128s.
// ---------------------------------------------------------------------------
template<int CIN, int KP>
__global__ __launch_bounds__(128)
void conv13_wmma(const float* __restrict__ in, const float* __restrict__ wl,
                 const float* __restrict__ wr, float* __restrict__ hout) {
  const int bn = blockIdx.x;
  const int b = bn / NIND, n = bn % NIND;
  const int tid = threadIdx.x;
  __shared__ __attribute__((aligned(16))) _Float16 sW[OUTC * KP];      // [o][k]
  __shared__ __attribute__((aligned(16))) _Float16 sBt[SS][KP + 8];    // [s][k]
  const float* w = (n < HALFN) ? wl : wr;
  const int KREAL = CIN * 3;

  for (int idx = tid; idx < OUTC * KP; idx += 128) {
    int o = idx / KP, k = idx % KP;
    float v = (k < KREAL) ? w[(size_t)o * KREAL + k] : 0.0f; // w[o][c][t], k=c*3+t
    sW[idx] = (_Float16)v;
  }
  for (int idx = tid; idx < KP * SS; idx += 128) {
    int k = idx / SS, s = idx % SS;       // consecutive tid -> consecutive s (coalesced)
    float v = 0.f;
    if (k < KREAL) {
      int c = k / 3, t = k - 3 * c;
      int ss = s + t - 1;
      if (ss >= 0 && ss < SS)
        v = in[((size_t)(b * CIN + c) * NIND + n) * SS + ss];
    }
    sBt[s][k] = (_Float16)v;
  }
  __syncthreads();

  const int wave = tid >> 5, lane = tid & 31;
  const int lm = lane & 15;
  const int hi = lane >> 4;                 // 0 or 1
  const int otile = wave & 1;               // fixed per wave -> A is loop-invariant
  const int sbase = wave >> 1;              // stiles: sbase + 2*tt
  const int arow = (otile * 16 + lm) * KP;

  // Preload all A fragments (ISA A 16-bit layout: elems 0..7 = K abase+0..7,
  // elems 8..15 = K abase+16..23; abase = chunk*32 + hi*8).
  v16h afrag[KP / 32];
#pragma unroll
  for (int chunk = 0; chunk < KP / 32; ++chunk) {
    const int abase = chunk * 32 + hi * 8;
    v8h alo = *(const v8h*)&sW[arow + abase];
    v8h ahi = *(const v8h*)&sW[arow + abase + 16];
    afrag[chunk] = __builtin_shufflevector(alo, ahi,
        0, 1, 2, 3, 4, 5, 6, 7, 8, 9, 10, 11, 12, 13, 14, 15);
  }

#pragma unroll
  for (int tt = 0; tt < 4; ++tt) {
    const int stile = sbase + tt * 2;
    const int s = stile * 16 + lm;
    v8f acc = {0.f, 0.f, 0.f, 0.f, 0.f, 0.f, 0.f, 0.f};
#pragma unroll
    for (int chunk = 0; chunk < KP / 32; ++chunk) {
      const int bb = chunk * 32 + hi * 8;   // B: K bb+0..7 and bb+16..23 (matches A)
      v8h blo = *(const v8h*)&sBt[s][bb];
      v8h bhi = *(const v8h*)&sBt[s][bb + 16];
      v16h bfr = __builtin_shufflevector(blo, bhi,
          0, 1, 2, 3, 4, 5, 6, 7, 8, 9, 10, 11, 12, 13, 14, 15);
      acc = __builtin_amdgcn_wmma_f32_16x16x32_f16(false, afrag[chunk], false, bfr,
                                                   (short)0, acc, false, false);
    }
#pragma unroll
    for (int r = 0; r < 8; ++r) {
      int o = otile * 16 + r + hi * 8;
      hout[((size_t)(b * OUTC + o) * NIND + n) * SS + s] = acc[r];
    }
  }
}

// ---------------------------------------------------------------------------
// Per-edge attention MLP: 8 -> 16 -> 32 -> 1 with LayerNorms, LeakyReLU, sigmoid
// ---------------------------------------------------------------------------
__global__ __launch_bounds__(128)
void att_mlp_kernel(const float* __restrict__ ea,
                    const float* __restrict__ bng, const float* __restrict__ bnb,
                    const float* __restrict__ w1, const float* __restrict__ b1,
                    const float* __restrict__ g1, const float* __restrict__ bb1,
                    const float* __restrict__ w2, const float* __restrict__ b2,
                    const float* __restrict__ g2, const float* __restrict__ bb2,
                    const float* __restrict__ w3, const float* __restrict__ b3,
                    float* __restrict__ att) {
  int e = blockIdx.x * blockDim.x + threadIdx.x;
  if (e >= EDGES) return;
  const float BNS = 0.99999500003749968f;  // 1/sqrt(1+1e-5)
  float a[8];
#pragma unroll
  for (int j = 0; j < 8; ++j)
    a[j] = ea[(size_t)e * 8 + j] * BNS * bng[j] + bnb[j];

  float h1[16];
  float m = 0.f;
#pragma unroll
  for (int o = 0; o < 16; ++o) {
    float s = b1[o];
#pragma unroll
    for (int j = 0; j < 8; ++j) s += a[j] * w1[o * 8 + j];
    h1[o] = s; m += s;
  }
  m *= (1.f / 16.f);
  float v = 0.f;
#pragma unroll
  for (int o = 0; o < 16; ++o) { float d = h1[o] - m; v += d * d; }
  float rstd = rsqrtf(v * (1.f / 16.f) + 1e-5f);
#pragma unroll
  for (int o = 0; o < 16; ++o)
    h1[o] = fmaxf((h1[o] - m) * rstd * g1[o] + bb1[o], 0.f);

  float h2[32];
  m = 0.f;
#pragma unroll
  for (int o = 0; o < 32; ++o) {
    float s = b2[o];
#pragma unroll
    for (int j = 0; j < 16; ++j) s += h1[j] * w2[o * 16 + j];
    h2[o] = s; m += s;
  }
  m *= (1.f / 32.f);
  v = 0.f;
#pragma unroll
  for (int o = 0; o < 32; ++o) { float d = h2[o] - m; v += d * d; }
  rstd = rsqrtf(v * (1.f / 32.f) + 1e-5f);
#pragma unroll
  for (int o = 0; o < 32; ++o)
    h2[o] = fmaxf((h2[o] - m) * rstd * g2[o] + bb2[o], 0.f);

  float z = b3[0];
#pragma unroll
  for (int j = 0; j < 32; ++j) z += h2[j] * w3[j];
  z = (z >= 0.f) ? z : 0.2f * z;        // LeakyReLU(0.2)
  att[e] = 1.f / (1.f + expf(-z));      // sigmoid
}

// ---------------------------------------------------------------------------
// xg = einsum('bcns,oc->bons', h, gw) + gb, stored in node-major [N][512] layout
// ---------------------------------------------------------------------------
__global__ __launch_bounds__(256)
void einsum_gcn(const float* __restrict__ h, const float* __restrict__ gw,
                const float* __restrict__ gb, float* __restrict__ nodes) {
  int idx = blockIdx.x * 256 + threadIdx.x;
  if (idx >= BB * NIND * SS) return;
  int s = idx % SS; int n = (idx / SS) % NIND; int b = idx / (SS * NIND);
  float a0 = gb[0], a1 = gb[1], a2 = gb[2], a3 = gb[3];
#pragma unroll 4
  for (int o = 0; o < 32; ++o) {
    float hv = h[((size_t)(b * OUTC + o) * NIND + n) * SS + s];
    a0 += hv * gw[o]; a1 += hv * gw[32 + o];
    a2 += hv * gw[64 + o]; a3 += hv * gw[96 + o];
  }
  size_t base = (size_t)(b * NIND + n) * FEAT + s;
  nodes[base] = a0; nodes[base + 128] = a1;
  nodes[base + 256] = a2; nodes[base + 384] = a3;
}

__global__ void zero_f32(float* __restrict__ p, int count) {
  int i = blockIdx.x * blockDim.x + threadIdx.x;
  if (i < count) p[i] = 0.f;
}

// ---------------------------------------------------------------------------
// Scatter-add:  aggr[dst[e]] += nodes[src[e]] * att[e].  One block per edge.
// ---------------------------------------------------------------------------
__global__ __launch_bounds__(128)
void scatter_edges(const float* __restrict__ nodes, const float* __restrict__ att,
                   const int* __restrict__ src, const int* __restrict__ dst,
                   float* __restrict__ aggr) {
  int e = blockIdx.x;
  float a = att[e];
  int sv = src[e], dv = dst[e];
  int f = threadIdx.x * 4;
  const float4 v = *(const float4*)(nodes + (size_t)sv * FEAT + f);
  float* o = aggr + (size_t)dv * FEAT + f;
  atomicAdd(o + 0, v.x * a);
  atomicAdd(o + 1, v.y * a);
  atomicAdd(o + 2, v.z * a);
  atomicAdd(o + 3, v.w * a);
}

// ---------------------------------------------------------------------------
// MessageNorm: aggr = aggr / max(||aggr||,1e-12) * ||nodes|| * mn_scale, in-place
// ---------------------------------------------------------------------------
__global__ __launch_bounds__(128)
void msgnorm(float* __restrict__ aggr, const float* __restrict__ nodes,
             const float* __restrict__ mn, int layer) {
  int vtx = blockIdx.x; int tid = threadIdx.x;
  __shared__ float sA[128], sN[128];
  __shared__ float sc;
  float ssa = 0.f, ssn = 0.f;
  size_t base = (size_t)vtx * FEAT;
#pragma unroll
  for (int j = 0; j < 4; ++j) {
    float va = aggr[base + tid + 128 * j];  ssa += va * va;
    float vn = nodes[base + tid + 128 * j]; ssn += vn * vn;
  }
  sA[tid] = ssa; sN[tid] = ssn; __syncthreads();
  for (int o = 64; o > 0; o >>= 1) {
    if (tid < o) { sA[tid] += sA[tid + o]; sN[tid] += sN[tid + o]; }
    __syncthreads();
  }
  if (tid == 0) {
    float na = sqrtf(sA[0]), xn = sqrtf(sN[0]);
    sc = xn * mn[layer] / fmaxf(na, 1e-12f);
  }
  __syncthreads();
  float s = sc;
#pragma unroll
  for (int j = 0; j < 4; ++j) aggr[base + tid + 128 * j] *= s;
}

// ---------------------------------------------------------------------------
// concat([gcn(4ch), h(32ch)]) (+prev) -> InstanceNorm over (n,s) -> xs;
// also writes relu+maxpool(1,2) slice of the final output.
// One block per (b,c).
// ---------------------------------------------------------------------------
__device__ __forceinline__ float fetch_val(const float* __restrict__ gcnout,
                                           const float* __restrict__ h,
                                           const float* __restrict__ prev,
                                           int b, int c, int n, int s) {
  float v;
  if (c < GCNC) v = gcnout[(size_t)(b * NIND + n) * FEAT + c * SS + s];
  else          v = h[((size_t)(b * OUTC + (c - GCNC)) * NIND + n) * SS + s];
  if (prev)     v += prev[((size_t)(b * CMID + c) * NIND + n) * SS + s];
  return v;
}

__global__ __launch_bounds__(256)
void combine_inorm(const float* __restrict__ gcnout, const float* __restrict__ h,
                   const float* __restrict__ prev, float* __restrict__ xsout,
                   float* __restrict__ out, int layer) {
  int b = blockIdx.x / CMID, c = blockIdx.x % CMID;
  int tid = threadIdx.x;
  __shared__ float s1[256], s2[256];
  __shared__ float smean, srstd;
  const int M = NIND * SS;
  float sum = 0.f, ssq = 0.f;
  for (int e = tid; e < M; e += 256) {
    int n = e >> 7, s = e & 127;
    float v = fetch_val(gcnout, h, prev, b, c, n, s);
    sum += v; ssq += v * v;
  }
  s1[tid] = sum; s2[tid] = ssq; __syncthreads();
  for (int o = 128; o > 0; o >>= 1) {
    if (tid < o) { s1[tid] += s1[tid + o]; s2[tid] += s2[tid + o]; }
    __syncthreads();
  }
  if (tid == 0) {
    float mean = s1[0] / (float)M;
    float var  = s2[0] / (float)M - mean * mean;
    smean = mean; srstd = rsqrtf(var + 1e-5f);
  }
  __syncthreads();
  float mean = smean, rstd = srstd;
  for (int p = tid; p < M / 2; p += 256) {
    int e0 = 2 * p; int n = e0 >> 7, s = e0 & 127;
    float v0 = (fetch_val(gcnout, h, prev, b, c, n, s)     - mean) * rstd;
    float v1 = (fetch_val(gcnout, h, prev, b, c, n, s + 1) - mean) * rstd;
    size_t xo = ((size_t)(b * CMID + c) * NIND + n) * SS + s;
    xsout[xo] = v0; xsout[xo + 1] = v1;
    out[((size_t)(b * (NLAY * CMID) + layer * CMID + c) * NIND + n) * (SS / 2) + (s >> 1)] =
        fmaxf(fmaxf(v0, 0.f), fmaxf(v1, 0.f));
  }
}

// ---------------------------------------------------------------------------
extern "C" void kernel_launch(void* const* d_in, const int* in_sizes, int n_in,
                              void* d_out, int out_size, void* d_ws, size_t ws_size,
                              hipStream_t stream) {
  (void)in_sizes; (void)n_in; (void)out_size; (void)ws_size;
  const float* x   = (const float*)d_in[0];
  const int*   ei  = (const int*)  d_in[1];
  const float* ea  = (const float*)d_in[2];
  const float* wl0 = (const float*)d_in[3];
  const float* wr0 = (const float*)d_in[4];
  const float* wl  = (const float*)d_in[5];
  const float* wr  = (const float*)d_in[6];
  const float* gw  = (const float*)d_in[7];
  const float* gb  = (const float*)d_in[8];
  const float* bng = (const float*)d_in[9];
  const float* bnb = (const float*)d_in[10];
  const float* w1  = (const float*)d_in[11];
  const float* b1  = (const float*)d_in[12];
  const float* g1  = (const float*)d_in[13];
  const float* bb1 = (const float*)d_in[14];
  const float* w2  = (const float*)d_in[15];
  const float* b2  = (const float*)d_in[16];
  const float* g2  = (const float*)d_in[17];
  const float* bb2 = (const float*)d_in[18];
  const float* w3  = (const float*)d_in[19];
  const float* b3  = (const float*)d_in[20];
  const float* mn  = (const float*)d_in[21];
  const int* src = ei;
  const int* dst = ei + EDGES;

  char* ws = (char*)d_ws;
  size_t off = 0;
  float* hbuf  = (float*)(ws + off); off += (size_t)BB * OUTC * NIND * SS * 4;   // 78.6 MB
  float* nodes = (float*)(ws + off); off += (size_t)NN * FEAT * 4;               // 9.8 MB
  float* aggr  = (float*)(ws + off); off += (size_t)NN * FEAT * 4;               // 9.8 MB
  float* attb  = (float*)(ws + off); off += (size_t)EDGES * 4;                   // 0.6 MB
  float* xsA   = (float*)(ws + off); off += (size_t)BB * CMID * NIND * SS * 4;   // 88.5 MB
  float* xsB   = (float*)(ws + off); off += (size_t)BB * CMID * NIND * SS * 4;   // 88.5 MB
  float* xs_prev = xsA;
  float* xs_cur  = xsB;
  float* outp = (float*)d_out;

  for (int i = 0; i < NLAY; ++i) {
    if (i == 0)
      conv13_wmma<CIN0, 64><<<BB * NIND, 128, 0, stream>>>(x, wl0, wr0, hbuf);
    else
      conv13_wmma<CMID, 128><<<BB * NIND, 128, 0, stream>>>(
          xs_prev, wl + (size_t)(i - 1) * OUTC * CMID * 3,
          wr + (size_t)(i - 1) * OUTC * CMID * 3, hbuf);

    att_mlp_kernel<<<EDGES / 128, 128, 0, stream>>>(
        ea, bng + i * 8, bnb + i * 8,
        w1 + i * 16 * 8, b1 + i * 16, g1 + i * 16, bb1 + i * 16,
        w2 + i * 32 * 16, b2 + i * 32, g2 + i * 32, bb2 + i * 32,
        w3 + i * 32, b3 + i, attb);

    einsum_gcn<<<(BB * NIND * SS + 255) / 256, 256, 0, stream>>>(
        hbuf, gw + i * GCNC * OUTC, gb + i * GCNC, nodes);

    zero_f32<<<(NN * FEAT + 255) / 256, 256, 0, stream>>>(aggr, NN * FEAT);

    scatter_edges<<<EDGES, 128, 0, stream>>>(nodes, attb, src, dst, aggr);

    msgnorm<<<NN, 128, 0, stream>>>(aggr, nodes, mn, i);

    combine_inorm<<<BB * CMID, 256, 0, stream>>>(
        aggr, hbuf, (i == 0) ? nullptr : xs_prev, xs_cur, outp, i);

    float* t = xs_prev; xs_prev = xs_cur; xs_cur = t;
  }
}
